// QuantLinear_69647189672510
// MI455X (gfx1250) — matrix operand order repaired and checked by reading, not compiled, not run on hardware
//
#include <hip/hip_runtime.h>

// ---- CDNA5 WMMA vector types ----
typedef __attribute__((ext_vector_type(16))) _Float16 v16h;
typedef __attribute__((ext_vector_type(2)))  _Float16 h2;
typedef __attribute__((ext_vector_type(8)))  float    v8f;

// Problem constants (fixed by the harness)
constexpr int M_DIM = 8192;
constexpr int K_DIM = 4096;
constexpr int N_DIM = 11008;
constexpr int GROUP = 128;

// Tiling
constexpr int BM = 256;
constexpr int BN = 128;
constexpr int BK = 64;
constexpr int LDA = BK + 8;   // 72 f16 = 144 bytes, multiple of 16B -> aligned b128 LDS ops
constexpr int LDB = BK + 8;

union FragAB {
    v16h  v;
    uint4 q[2];
};

// fp16 magic: 0x6400 == 1024.0; (0x6400 | q) == 1024 + q exactly for q in [0,15]
constexpr unsigned MAGIC2 = 0x64006400u;

// ---------- helpers ----------
__device__ __forceinline__ uint4 dequant_word(unsigned w, h2 spk, h2 subpk) {
    // pairs (q_p, q_{p+4}) -> k' positions (2p, 2p+1); K order within each
    // 8-chunk is the fixed permutation [0,4,1,5,2,6,3,7].
    uint4 o;
    h2 h0 = __builtin_bit_cast(h2, (w & 0x000F000Fu) | MAGIC2);
    h2 h1 = __builtin_bit_cast(h2, ((w >> 4) & 0x000F000Fu) | MAGIC2);
    h2 h2v = __builtin_bit_cast(h2, ((w >> 8) & 0x000F000Fu) | MAGIC2);
    h2 h3 = __builtin_bit_cast(h2, ((w >> 12) & 0x000F000Fu) | MAGIC2);
    o.x = __builtin_bit_cast(unsigned, (h2)(h0 * spk - subpk));
    o.y = __builtin_bit_cast(unsigned, (h2)(h1 * spk - subpk));
    o.z = __builtin_bit_cast(unsigned, (h2)(h2v * spk - subpk));
    o.w = __builtin_bit_cast(unsigned, (h2)(h3 * spk - subpk));
    return o;
}

// async-to-LDS builtin takes int4 pointers: (v4i AS(1)* src, v4i AS(3)* dst, imm, imm)
typedef int v4i_vs __attribute__((vector_size(16)));
typedef __attribute__((address_space(1))) v4i_vs* g1_v4i;
typedef __attribute__((address_space(3))) v4i_vs* l3_v4i;

__device__ __forceinline__ void async_copy16(const _Float16* g, _Float16* s) {
#if __has_builtin(__builtin_amdgcn_global_load_async_to_lds_b128)
    __builtin_amdgcn_global_load_async_to_lds_b128((g1_v4i)g, (l3_v4i)s, 0, 0);
#else
    *reinterpret_cast<uint4*>(s) = *reinterpret_cast<const uint4*>(g);
#endif
}

__device__ __forceinline__ void wait_async() {
#if __has_builtin(__builtin_amdgcn_s_wait_asynccnt)
    __builtin_amdgcn_s_wait_asynccnt(0);
#elif __has_builtin(__builtin_amdgcn_global_load_async_to_lds_b128)
    asm volatile("s_wait_asynccnt 0x0" ::: "memory");
#endif
}

// ---------- pass 1: x fp32 -> f16, sigma-permuted pairs ----------
__global__ __launch_bounds__(256)
void convert_x16(const float* __restrict__ x, _Float16* __restrict__ x16) {
    const size_t idx = (size_t)blockIdx.x * 256 + threadIdx.x;   // one 8-chunk
    const float* xp = x + idx * 8;
    const float4 fa = *reinterpret_cast<const float4*>(xp);
    const float4 fb = *reinterpret_cast<const float4*>(xp + 4);
    uint4 o;
    o.x = __builtin_bit_cast(unsigned, __builtin_amdgcn_cvt_pkrtz(fa.x, fb.x));
    o.y = __builtin_bit_cast(unsigned, __builtin_amdgcn_cvt_pkrtz(fa.y, fb.y));
    o.z = __builtin_bit_cast(unsigned, __builtin_amdgcn_cvt_pkrtz(fa.z, fb.z));
    o.w = __builtin_bit_cast(unsigned, __builtin_amdgcn_cvt_pkrtz(fa.w, fb.w));
    *reinterpret_cast<uint4*>(x16 + idx * 8) = o;
}

// ---------- pass 2: int4 -> f16 [N][K] (K contiguous, sigma-permuted) ----------
__global__ __launch_bounds__(256)
void dequant_w16(const int*   __restrict__ qweight,
                 const int*   __restrict__ qzeros,
                 const float* __restrict__ scales,
                 _Float16*    __restrict__ w16) {
    const int n = blockIdx.x * 256 + threadIdx.x;   // output column
    const int g = blockIdx.y;                       // 16 qweight rows = 1 group of 128 K
    const int z = (qzeros[(size_t)g * (N_DIM / 8) + (n >> 3)] >> (4 * (n & 7))) & 15;
    const float s = scales[(size_t)g * N_DIM + n];
    const _Float16 hs   = (_Float16)s;
    const _Float16 hsub = (_Float16)(s * (1024.0f + (float)z));
    const h2 spk   = {hs, hs};
    const h2 subpk = {hsub, hsub};
    _Float16* dst = w16 + (size_t)n * K_DIM + g * GROUP;   // 256B contiguous per thread
#pragma unroll
    for (int i = 0; i < 16; ++i) {
        const unsigned w = (unsigned)qweight[(size_t)(g * 16 + i) * N_DIM + n];  // coalesced over n
        *reinterpret_cast<uint4*>(dst + i * 8) = dequant_word(w, spk, subpk);
    }
}

// ---------- main GEMM ----------
// PRE=true : double-buffered software pipeline; A/B tiles preconverted f16,
//            staged with global_load_async_to_lds_b128 overlapping the WMMAs.
// PRE=false: in-kernel conversion fallback (used only if ws is too small).
template <bool PRE>
__global__ __launch_bounds__(256, 1)
void quantlinear_wmma_f16(const float*    __restrict__ x,
                          const int*      __restrict__ qweight,
                          const int*      __restrict__ qzeros,
                          const float*    __restrict__ scales,
                          const float*    __restrict__ bias,
                          float*          __restrict__ out,
                          const _Float16* __restrict__ x16,
                          const _Float16* __restrict__ w16) {
    __shared__ _Float16 Alds[2][BM * LDA];   // [buf][m][k']
    __shared__ _Float16 Blds[2][BN * LDB];   // [buf][n][k']  (K contiguous per n)

    const int tid  = threadIdx.x;
    const int wave = tid >> 5;
    const int lane = tid & 31;
    const int lm    = lane & 15;
    const int lhalf = lane >> 4;

    const int waveM = wave & 3;    // 4 row-groups of 64 rows
    const int waveN = wave >> 2;   // 2 col-groups of 64 cols

    const int n0 = blockIdx.x * BN;
    const int m0 = blockIdx.y * BM;

    v8f acc[4][4];
#pragma unroll
    for (int i = 0; i < 4; ++i)
#pragma unroll
        for (int j = 0; j < 4; ++j)
            acc[i][j] = (v8f){0.f, 0.f, 0.f, 0.f, 0.f, 0.f, 0.f, 0.f};

    // issue async copies for one K-tile into buffer `buf`
    auto stage_async = [&](int k0, int buf) {
        // A: 2048 16B chunks, 8 per thread
#pragma unroll
        for (int i = 0; i < 8; ++i) {
            const int cidx = i * 256 + tid;
            const int r    = cidx >> 3;
            const int c8   = cidx & 7;
            async_copy16(x16 + (size_t)(m0 + r) * K_DIM + k0 + c8 * 8,
                         &Alds[buf][r * LDA + c8 * 8]);
        }
        // B: 1024 16B chunks, 4 per thread
#pragma unroll
        for (int i = 0; i < 4; ++i) {
            const int cidx = i * 256 + tid;
            const int n    = cidx >> 3;
            const int c8   = cidx & 7;
            async_copy16(w16 + (size_t)(n0 + n) * K_DIM + k0 + c8 * 8,
                         &Blds[buf][n * LDB + c8 * 8]);
        }
    };

    // 32 WMMAs on one staged K-tile
    auto compute_tile = [&](int buf) {
#pragma unroll
        for (int ks = 0; ks < BK; ks += 32) {
            FragAB a[4], b[4];
#pragma unroll
            for (int tm = 0; tm < 4; ++tm) {
                const int m = waveM * 64 + tm * 16 + lm;
                const uint4* ap = reinterpret_cast<const uint4*>(
                    &Alds[buf][m * LDA + ks + lhalf * 8]);
                a[tm].q[0] = ap[0];   // K ks..      (16B)
                a[tm].q[1] = ap[2];   // K ks+16..   (+32B)
            }
#pragma unroll
            for (int tn = 0; tn < 4; ++tn) {
                const int n = waveN * 64 + tn * 16 + lm;
                const uint4* bp = reinterpret_cast<const uint4*>(
                    &Blds[buf][n * LDB + ks + lhalf * 8]);
                b[tn].q[0] = bp[0];
                b[tn].q[1] = bp[2];
            }
#pragma unroll
            for (int tm = 0; tm < 4; ++tm)
#pragma unroll
                for (int tn = 0; tn < 4; ++tn)
                    acc[tm][tn] = __builtin_amdgcn_wmma_f32_16x16x32_f16(
                        false, a[tm].v, false, b[tn].v,
                        (short)0, acc[tm][tn], false, false);
        }
    };

    if constexpr (PRE) {
        // ---- double-buffered pipeline: DMA for tile k+1 overlaps WMMAs on tile k ----
        int p = 0;
        stage_async(0, 0);
        for (int k0 = 0; k0 < K_DIM; k0 += BK) {
            wait_async();        // own copies for tile k complete (issued last iteration)
            __syncthreads();     // all waves' copies landed; all done reading other buffer
            if (k0 + BK < K_DIM)
                stage_async(k0 + BK, p ^ 1);   // background DMA during compute
            compute_tile(p);
            p ^= 1;
        }
    } else {
        // ---- fallback: in-kernel conversion, single buffer ----
        const int brr  = tid >> 5;
        const int bnn0 = (tid & 31) * 4;
        for (int k0 = 0; k0 < K_DIM; k0 += BK) {
#pragma unroll
            for (int i = 0; i < 8; ++i) {
                const int cidx = i * 256 + tid;
                const int r    = cidx >> 3;
                const int c8   = cidx & 7;
                const float* xp = x + (size_t)(m0 + r) * K_DIM + k0 + c8 * 8;
                const float4 fa = *reinterpret_cast<const float4*>(xp);
                const float4 fb = *reinterpret_cast<const float4*>(xp + 4);
                uint4 o;
                o.x = __builtin_bit_cast(unsigned, __builtin_amdgcn_cvt_pkrtz(fa.x, fb.x));
                o.y = __builtin_bit_cast(unsigned, __builtin_amdgcn_cvt_pkrtz(fa.y, fb.y));
                o.z = __builtin_bit_cast(unsigned, __builtin_amdgcn_cvt_pkrtz(fa.z, fb.z));
                o.w = __builtin_bit_cast(unsigned, __builtin_amdgcn_cvt_pkrtz(fa.w, fb.w));
                *reinterpret_cast<uint4*>(&Alds[0][r * LDA + c8 * 8]) = o;
            }
            const int g = k0 / GROUP;
            const size_t qrow = (size_t)(k0 / 8 + brr) * N_DIM + n0 + bnn0;
            const int4   w4 = *reinterpret_cast<const int4*>(qweight + qrow);
            const float4 s4 = *reinterpret_cast<const float4*>(scales + (size_t)g * N_DIM + n0 + bnn0);
            const int zw = qzeros[(size_t)g * (N_DIM / 8) + (n0 + bnn0) / 8];
            const int zbase = 4 * (bnn0 & 7);
            const unsigned wq[4] = {(unsigned)w4.x, (unsigned)w4.y, (unsigned)w4.z, (unsigned)w4.w};
            const float    sc[4] = {s4.x, s4.y, s4.z, s4.w};
#pragma unroll
            for (int j = 0; j < 4; ++j) {
                const int      z = (zw >> (zbase + 4 * j)) & 15;
                const float    s = sc[j];
                const _Float16 hs   = (_Float16)s;
                const _Float16 hsub = (_Float16)(s * (1024.0f + (float)z));
                const h2 spk   = {hs, hs};
                const h2 subpk = {hsub, hsub};
                *reinterpret_cast<uint4*>(&Blds[0][(bnn0 + j) * LDB + brr * 8]) =
                    dequant_word(wq[j], spk, subpk);
            }
            __syncthreads();
            compute_tile(0);
            __syncthreads();
        }
    }

    // ---------- Epilogue: D layout VGPR v -> M = v + 8*lhalf, N = lane&15 ----------
#pragma unroll
    for (int tn = 0; tn < 4; ++tn) {
        const int col = n0 + waveN * 64 + tn * 16 + lm;
        const float bv = bias[col];
#pragma unroll
        for (int tm = 0; tm < 4; ++tm) {
            const int rowbase = m0 + waveM * 64 + tm * 16 + lhalf * 8;
#pragma unroll
            for (int v = 0; v < 8; ++v) {
                out[(size_t)(rowbase + v) * N_DIM + col] = acc[tm][tn][v] + bv;
            }
        }
    }
}

extern "C" void kernel_launch(void* const* d_in, const int* in_sizes, int n_in,
                              void* d_out, int out_size, void* d_ws, size_t ws_size,
                              hipStream_t stream) {
    const float* x       = (const float*)d_in[0];
    const int*   qweight = (const int*)  d_in[1];
    const int*   qzeros  = (const int*)  d_in[2];
    const float* scales  = (const float*)d_in[3];
    const float* bias    = (const float*)d_in[4];
    float*       out     = (float*)d_out;

    const size_t needA = (size_t)M_DIM * K_DIM * sizeof(_Float16);   // 64 MiB
    const size_t needB = (size_t)N_DIM * K_DIM * sizeof(_Float16);   // 86 MiB

    dim3 grid(N_DIM / BN, M_DIM / BM);   // 86 x 32
    dim3 block(256);

    if (ws_size >= needA + needB) {
        _Float16* x16 = (_Float16*)d_ws;
        _Float16* w16 = (_Float16*)((char*)d_ws + needA);
        convert_x16<<<(int)(((size_t)M_DIM * K_DIM / 8) / 256), 256, 0, stream>>>(x, x16);
        dequant_w16<<<dim3(N_DIM / 256, K_DIM / GROUP), 256, 0, stream>>>(qweight, qzeros, scales, w16);
        quantlinear_wmma_f16<true><<<grid, block, 0, stream>>>(
            x, qweight, qzeros, scales, bias, out, x16, w16);
    } else {
        quantlinear_wmma_f16<false><<<grid, block, 0, stream>>>(
            x, qweight, qzeros, scales, bias, out, nullptr, nullptr);
    }
}
